// TransformerBlock_26044681683553
// MI455X (gfx1250) — compile-verified, matrix operand order checked by reading, not run on hardware
//
#include <hip/hip_runtime.h>
#include <hip/hip_bf16.h>

#define TB_E   1024
#define TB_H   16
#define TB_HD  64
#define TB_T   2048
#define TB_B   2
#define TB_M   (TB_B * TB_T)   /* 4096 rows */
#define TB_F   (4 * TB_E)      /* 4096 ffn dim */

typedef __bf16 bf16_t;
typedef __attribute__((ext_vector_type(16))) __bf16 bf16x16;
typedef __attribute__((ext_vector_type(8)))  __bf16 bf16x8;
typedef __attribute__((ext_vector_type(8)))  float  f32x8;

// Load a 16x32 bf16 WMMA A/B fragment for this lane.
// Per ISA: lane holds row/col (lane&15); K elems = 8*(lane>>4) + (j<8?0:16) + (j&7)
// i.e. two contiguous runs of 8 bf16 at +0 and +16 from (base + 8*(lane>>4)).
__device__ __forceinline__ bf16x16 load_frag16(const bf16_t* p) {
    bf16x8 lo = *(const bf16x8*)(p);
    bf16x8 hi = *(const bf16x8*)(p + 16);
    bf16x16 r;
#pragma unroll
    for (int i = 0; i < 8; ++i) { r[i] = lo[i]; r[i + 8] = hi[i]; }
    return r;
}

// ---------------------------------------------------------------------------
// Weight convert + transpose: Wt[n*K + k] = bf16(W[k*N + n])
// ---------------------------------------------------------------------------
__global__ void w_transpose_bf16(const float* __restrict__ W,
                                 bf16_t* __restrict__ Wt, int K, int N) {
    size_t idx = (size_t)blockIdx.x * blockDim.x + threadIdx.x;
    if (idx >= (size_t)K * N) return;
    int k = (int)(idx % K);
    int n = (int)(idx / K);
    Wt[idx] = (bf16_t)W[(size_t)k * N + n];
}

// V: [B][T][H*HD] bf16  ->  Vt: [B][H][HD][T] bf16 (t fastest: coalesced writes)
__global__ void v_transpose_bf16(const bf16_t* __restrict__ V,
                                 bf16_t* __restrict__ Vt) {
    size_t idx = (size_t)blockIdx.x * blockDim.x + threadIdx.x;
    if (idx >= (size_t)TB_B * TB_T * TB_E) return;
    int t = (int)(idx % TB_T);
    size_t r = idx / TB_T;
    int hd = (int)(r % TB_HD); r /= TB_HD;
    int h  = (int)(r % TB_H);
    int b  = (int)(r / TB_H);
    Vt[idx] = V[((size_t)(b * TB_T + t)) * TB_E + h * TB_HD + hd];
}

// ---------------------------------------------------------------------------
// LayerNorm (fp32 in, bf16 out): one block per row
// ---------------------------------------------------------------------------
__global__ __launch_bounds__(256) void layernorm_bf16(
    const float* __restrict__ x, const float* __restrict__ g,
    const float* __restrict__ be, bf16_t* __restrict__ out, int E) {
    __shared__ float red[16];
    int row = blockIdx.x;
    const float* xr = x + (size_t)row * E;
    float s = 0.f, ss = 0.f;
    for (int i = threadIdx.x; i < E; i += blockDim.x) {
        float v = xr[i]; s += v; ss += v * v;
    }
#pragma unroll
    for (int o = 16; o > 0; o >>= 1) {
        s  += __shfl_xor(s,  o, 32);
        ss += __shfl_xor(ss, o, 32);
    }
    int wid = threadIdx.x >> 5;
    if ((threadIdx.x & 31) == 0) { red[wid] = s; red[8 + wid] = ss; }
    __syncthreads();
    float ts = 0.f, tss = 0.f;
#pragma unroll
    for (int i = 0; i < 8; ++i) { ts += red[i]; tss += red[8 + i]; }
    float mu   = ts / (float)E;
    float var  = tss / (float)E - mu * mu;
    float rstd = rsqrtf(var + 1e-5f);
    for (int i = threadIdx.x; i < E; i += blockDim.x)
        out[(size_t)row * E + i] = (bf16_t)((xr[i] - mu) * rstd * g[i] + be[i]);
}

// ---------------------------------------------------------------------------
// bf16 WMMA GEMM: C[M][N] = A[M][K] * Bt[N][K]^T (+ epilogue)
// block = 256 thr = 8 waves arranged 2(M) x 4(N); wave tile 32x64 (2x4 WMMA)
// block tile 64x256. K multiple of 64; two-stage ping-pong fragment pipeline.
// ---------------------------------------------------------------------------
enum { EPI_BF16 = 0, EPI_RES_F32 = 1, EPI_GELU_BF16 = 2 };

struct FragPtrs { const bf16_t* a[2]; const bf16_t* b[4]; };

__device__ __forceinline__ void load_set(const FragPtrs& fp, int k,
                                         bf16x16* af, bf16x16* bfv) {
#pragma unroll
    for (int i = 0; i < 2; ++i) af[i]  = load_frag16(fp.a[i] + k);
#pragma unroll
    for (int i = 0; i < 4; ++i) bfv[i] = load_frag16(fp.b[i] + k);
}

template <int EPI>
__global__ __launch_bounds__(256) void gemm_bf16_wmma(
    const bf16_t* __restrict__ A, const bf16_t* __restrict__ Bt,
    const float* __restrict__ bias, const float* __restrict__ res,
    float* __restrict__ outF, bf16_t* __restrict__ outB,
    int M, int N, int K) {
    const int lane = threadIdx.x & 31;
    const int hf   = lane >> 4;
    const int ln   = lane & 15;
    const int wave = threadIdx.x >> 5;
    const int rowBase = blockIdx.x * 64  + (wave & 1) * 32;
    const int colBase = blockIdx.y * 256 + (wave >> 1) * 64;

    f32x8 acc[2][4];
#pragma unroll
    for (int i = 0; i < 2; ++i)
#pragma unroll
        for (int j = 0; j < 4; ++j)
#pragma unroll
            for (int r = 0; r < 8; ++r) acc[i][j][r] = 0.f;

    FragPtrs fp;
#pragma unroll
    for (int i = 0; i < 2; ++i)
        fp.a[i] = A + (size_t)(rowBase + i * 16 + ln) * K + hf * 8;
#pragma unroll
    for (int i = 0; i < 4; ++i)
        fp.b[i] = Bt + (size_t)(colBase + i * 16 + ln) * K + hf * 8;

    bf16x16 aS0[2], bS0[4], aS1[2], bS1[4];
    load_set(fp, 0, aS0, bS0);                       // prologue: set0 @ k=0
    for (int k0 = 0; k0 < K; k0 += 64) {
        load_set(fp, k0 + 32, aS1, bS1);             // prefetch set1
#pragma unroll
        for (int mi = 0; mi < 2; ++mi)
#pragma unroll
            for (int ni = 0; ni < 4; ++ni)
                acc[mi][ni] = __builtin_amdgcn_wmma_f32_16x16x32_bf16(
                    false, aS0[mi], false, bS0[ni], (short)0, acc[mi][ni], false, false);
        if (k0 + 64 < K) load_set(fp, k0 + 64, aS0, bS0);  // prefetch set0
#pragma unroll
        for (int mi = 0; mi < 2; ++mi)
#pragma unroll
            for (int ni = 0; ni < 4; ++ni)
                acc[mi][ni] = __builtin_amdgcn_wmma_f32_16x16x32_bf16(
                    false, aS1[mi], false, bS1[ni], (short)0, acc[mi][ni], false, false);
    }

#pragma unroll
    for (int mi = 0; mi < 2; ++mi)
#pragma unroll
        for (int ni = 0; ni < 4; ++ni)
#pragma unroll
            for (int r = 0; r < 8; ++r) {
                int gm = rowBase + mi * 16 + r + 8 * hf;   // C layout: M = reg + 8*half
                int gn = colBase + ni * 16 + ln;           //           N = lane&15
                float v = acc[mi][ni][r];
                size_t o = (size_t)gm * N + gn;
                if (EPI == EPI_BF16) {
                    outB[o] = (bf16_t)v;
                } else if (EPI == EPI_RES_F32) {
                    outF[o] = v + bias[gn] + res[o];
                } else {  // exact GELU
                    float u = v + bias[gn];
                    outB[o] = (bf16_t)(0.5f * u * (1.0f + erff(u * 0.7071067811865476f)));
                }
            }
}

// ---------------------------------------------------------------------------
// Flash attention, wave32 WMMA. One 16-query tile per wave, 4 waves/block.
// Computes S^T = K_blk * Q^T so per-query softmax stats are per-lane and the
// exponentiated P^T (C layout) is directly a legal B fragment for
// O^T += V^T * P^T (no LDS transpose).
// Q,K: [B*T][E] bf16 ; Vt: [B][H][HD][T] bf16 ; O: [B*T][E] bf16
// ---------------------------------------------------------------------------
template <bool MASKED>
__device__ __forceinline__ void attn_key_block(
    int kb, int qi, const bf16_t* kRow, const bf16_t* vRow,
    const bf16x16& qf0, const bf16x16& qf1,
    float& m_i, float& l_i, f32x8* oacc, int ln, int hf) {
    const float scale = 0.03125f;      // 1/sqrt(E) = 1/32 (ref scales by C^-0.5)
    // --- issue ALL fragment loads first (K for S, V for O: independent) ---
    bf16x16 kf[2][2], vf[4];
#pragma unroll
    for (int kt = 0; kt < 2; ++kt) {
        const bf16_t* kp = kRow + (size_t)(kb + kt * 16) * TB_E;
        kf[kt][0] = load_frag16(kp);
        kf[kt][1] = load_frag16(kp + 32);
    }
#pragma unroll
    for (int mt = 0; mt < 4; ++mt)
        vf[mt] = load_frag16(vRow + (size_t)(mt * 16) * TB_T + kb);
    // --- S^T tiles (2x 16keys x 16queries), contraction over HD=64 ---
    f32x8 st[2];
#pragma unroll
    for (int kt = 0; kt < 2; ++kt) {
        f32x8 c;
#pragma unroll
        for (int r = 0; r < 8; ++r) c[r] = 0.f;
        c = __builtin_amdgcn_wmma_f32_16x16x32_bf16(false, kf[kt][0], false, qf0, (short)0, c, false, false);
        c = __builtin_amdgcn_wmma_f32_16x16x32_bf16(false, kf[kt][1], false, qf1, (short)0, c, false, false);
        st[kt] = c;
    }
    // --- scale (+ causal mask only in tail block); per-query running max ---
    float sv[2][8];
    float rmax = -INFINITY;
#pragma unroll
    for (int kt = 0; kt < 2; ++kt)
#pragma unroll
        for (int r = 0; r < 8; ++r) {
            float v = st[kt][r] * scale;
            if (MASKED) {
                int key = kb + kt * 16 + r + 8 * hf;
                v = (key <= qi) ? v : -INFINITY;
            }
            sv[kt][r] = v;
            rmax = fmaxf(rmax, v);
        }
    rmax = fmaxf(rmax, __shfl_xor(rmax, 16, 32));   // combine both key halves
    float mnew  = fmaxf(m_i, rmax);
    float alpha = __expf(m_i - mnew);
    float rsum = 0.f;
    bf16x16 pf;
#pragma unroll
    for (int kt = 0; kt < 2; ++kt)
#pragma unroll
        for (int r = 0; r < 8; ++r) {
            float e = __expf(sv[kt][r] - mnew);
            rsum += e;
            pf[kt * 8 + r] = (bf16_t)e;   // P^T C-layout == B fragment
        }
    rsum += __shfl_xor(rsum, 16, 32);
    l_i = l_i * alpha + rsum;
    m_i = mnew;
    // --- O^T += V^T * P^T (4 hd tiles, contraction over 32 keys) ---
#pragma unroll
    for (int mt = 0; mt < 4; ++mt) {
        oacc[mt] = oacc[mt] * alpha;
        oacc[mt] = __builtin_amdgcn_wmma_f32_16x16x32_bf16(
            false, vf[mt], false, pf, (short)0, oacc[mt], false, false);
    }
}

#define ATTN_WAVES 4
__global__ __launch_bounds__(32 * ATTN_WAVES) void attn_flash_wmma(
    const bf16_t* __restrict__ Q, const bf16_t* __restrict__ Km,
    const bf16_t* __restrict__ Vt, bf16_t* __restrict__ O) {
    const int lane = threadIdx.x & 31;
    const int hf   = lane >> 4;
    const int ln   = lane & 15;
    const int gw   = blockIdx.x * ATTN_WAVES + (threadIdx.x >> 5);
    const int QT   = TB_T / 16;
    const int qt   = gw % QT;
    const int bh   = gw / QT;
    const int head = bh % TB_H;
    const int b    = bh / TB_H;
    const int qBase = qt * 16;
    const int qi    = qBase + ln;      // this lane's query

    // Persistent Q^T B-fragments (col = query = ln, K = hd)
    const bf16_t* qp = Q + ((size_t)(b * TB_T + qBase + ln)) * TB_E + head * TB_HD + hf * 8;
    bf16x16 qf0 = load_frag16(qp);
    bf16x16 qf1 = load_frag16(qp + 32);

    // per-lane row bases for K fragments and V^T fragments
    const bf16_t* kRow = Km + ((size_t)(b * TB_T + ln)) * TB_E + head * TB_HD + hf * 8;
    const bf16_t* vRow = Vt + ((size_t)(b * TB_H + head) * TB_HD + ln) * TB_T + hf * 8;

    float m_i = -INFINITY, l_i = 0.f;
    f32x8 oacc[4];
#pragma unroll
    for (int t = 0; t < 4; ++t)
#pragma unroll
        for (int r = 0; r < 8; ++r) oacc[t][r] = 0.f;

    // full (unmasked) key blocks: keys kb..kb+31 all < qBase <= every query
    const int nFull = qBase >> 5;
    for (int i = 0; i < nFull; ++i)
        attn_key_block<false>(i * 32, qi, kRow, vRow, qf0, qf1, m_i, l_i, oacc, ln, hf);
    // single masked tail block covers keys up to qBase+15
    attn_key_block<true>(nFull * 32, qi, kRow, vRow, qf0, qf1, m_i, l_i, oacc, ln, hf);

    // --- normalize and write: lane owns one query, regs span hd ---
    float rinv = 1.f / l_i;
    bf16_t* op = O + ((size_t)(b * TB_T + qi)) * TB_E + head * TB_HD;
#pragma unroll
    for (int mt = 0; mt < 4; ++mt)
#pragma unroll
        for (int r = 0; r < 8; ++r)
            op[mt * 16 + r + 8 * hf] = (bf16_t)(oacc[mt][r] * rinv);
}

// ---------------------------------------------------------------------------
// Host-side orchestration
// ---------------------------------------------------------------------------
static inline size_t al256(size_t x) { return (x + 255) & ~(size_t)255; }

extern "C" void kernel_launch(void* const* d_in, const int* in_sizes, int n_in,
                              void* d_out, int out_size, void* d_ws, size_t ws_size,
                              hipStream_t stream) {
    const float* x   = (const float*)d_in[0];
    const float* Wq  = (const float*)d_in[1];
    const float* Wk  = (const float*)d_in[2];
    const float* Wv  = (const float*)d_in[3];
    const float* Wp  = (const float*)d_in[4];
    const float* bp  = (const float*)d_in[5];
    const float* W1  = (const float*)d_in[6];
    const float* b1  = (const float*)d_in[7];
    const float* W2  = (const float*)d_in[8];
    const float* b2  = (const float*)d_in[9];
    const float* g1  = (const float*)d_in[10];
    const float* be1 = (const float*)d_in[11];
    const float* g2  = (const float*)d_in[12];
    const float* be2 = (const float*)d_in[13];

    // workspace carve-up
    char* p = (char*)d_ws;
    const size_t szEE  = (size_t)TB_E * TB_E;       // 1M
    const size_t szEF  = (size_t)TB_E * TB_F;       // 4M
    const size_t szME  = (size_t)TB_M * TB_E;       // 4M
    const size_t szMF  = (size_t)TB_M * TB_F;       // 16M
    bf16_t* wq_t = (bf16_t*)p; p += al256(szEE * 2);
    bf16_t* wk_t = (bf16_t*)p; p += al256(szEE * 2);
    bf16_t* wv_t = (bf16_t*)p; p += al256(szEE * 2);
    bf16_t* wp_t = (bf16_t*)p; p += al256(szEE * 2);
    bf16_t* w1_t = (bf16_t*)p; p += al256(szEF * 2);
    bf16_t* w2_t = (bf16_t*)p; p += al256(szEF * 2);
    bf16_t* h1   = (bf16_t*)p; p += al256(szME * 2);
    bf16_t* qb   = (bf16_t*)p; p += al256(szME * 2);
    bf16_t* kbuf = (bf16_t*)p; p += al256(szME * 2);
    bf16_t* vbuf = (bf16_t*)p; p += al256(szME * 2);
    bf16_t* vt   = (bf16_t*)p; p += al256(szME * 2);
    bf16_t* ao   = (bf16_t*)p; p += al256(szME * 2);
    float*  x1   = (float*)p;  p += al256(szME * 4);
    bf16_t* h2   = (bf16_t*)p; p += al256(szME * 2);
    bf16_t* ff1  = (bf16_t*)p; p += al256(szMF * 2);

    // 1) weights -> bf16, transposed to [N][K]
    w_transpose_bf16<<<(unsigned)((szEE + 255) / 256), 256, 0, stream>>>(Wq, wq_t, TB_E, TB_E);
    w_transpose_bf16<<<(unsigned)((szEE + 255) / 256), 256, 0, stream>>>(Wk, wk_t, TB_E, TB_E);
    w_transpose_bf16<<<(unsigned)((szEE + 255) / 256), 256, 0, stream>>>(Wv, wv_t, TB_E, TB_E);
    w_transpose_bf16<<<(unsigned)((szEE + 255) / 256), 256, 0, stream>>>(Wp, wp_t, TB_E, TB_E);
    w_transpose_bf16<<<(unsigned)((szEF + 255) / 256), 256, 0, stream>>>(W1, w1_t, TB_E, TB_F);
    w_transpose_bf16<<<(unsigned)((szEF + 255) / 256), 256, 0, stream>>>(W2, w2_t, TB_F, TB_E);

    // 2) LN1
    layernorm_bf16<<<TB_M, 256, 0, stream>>>(x, g1, be1, h1, TB_E);

    // 3) QKV GEMMs  (block tile 64x256)
    dim3 gEE(TB_M / 64, TB_E / 256);
    gemm_bf16_wmma<EPI_BF16><<<gEE, 256, 0, stream>>>(h1, wq_t, nullptr, nullptr, nullptr, qb,   TB_M, TB_E, TB_E);
    gemm_bf16_wmma<EPI_BF16><<<gEE, 256, 0, stream>>>(h1, wk_t, nullptr, nullptr, nullptr, kbuf, TB_M, TB_E, TB_E);
    gemm_bf16_wmma<EPI_BF16><<<gEE, 256, 0, stream>>>(h1, wv_t, nullptr, nullptr, nullptr, vbuf, TB_M, TB_E, TB_E);

    // 4) V -> [B][H][HD][T]
    v_transpose_bf16<<<(unsigned)((szME + 255) / 256), 256, 0, stream>>>(vbuf, vt);

    // 5) flash attention (B*H*(T/16) = 4096 wave-tiles, 4 per block)
    attn_flash_wmma<<<(TB_B * TB_H * (TB_T / 16)) / ATTN_WAVES, 32 * ATTN_WAVES, 0, stream>>>(qb, kbuf, vt, ao);

    // 6) proj + bias + residual -> x1 (f32)
    gemm_bf16_wmma<EPI_RES_F32><<<gEE, 256, 0, stream>>>(ao, wp_t, bp, x, x1, nullptr, TB_M, TB_E, TB_E);

    // 7) LN2
    layernorm_bf16<<<TB_M, 256, 0, stream>>>(x1, g2, be2, h2, TB_E);

    // 8) FFN up + GELU
    dim3 gEF(TB_M / 64, TB_F / 256);
    gemm_bf16_wmma<EPI_GELU_BF16><<<gEF, 256, 0, stream>>>(h2, w1_t, b1, nullptr, nullptr, ff1, TB_M, TB_F, TB_E);

    // 9) FFN down + bias + residual -> d_out (f32)
    gemm_bf16_wmma<EPI_RES_F32><<<gEE, 256, 0, stream>>>(ff1, w2_t, b2, x1, (float*)d_out, nullptr, TB_M, TB_E, TB_F);

    (void)in_sizes; (void)n_in; (void)out_size; (void)ws_size;
}
